// ProfileMatchingGNN_15685220565283
// MI455X (gfx1250) — matrix operand order repaired and checked by reading, not compiled, not run on hardware
//
#include <hip/hip_runtime.h>
#include <hip/hip_fp16.h>

typedef __attribute__((ext_vector_type(16))) _Float16 v16h;
typedef __attribute__((ext_vector_type(8)))  _Float16 v8h;
typedef __attribute__((ext_vector_type(8)))  float    v8f;
typedef __attribute__((ext_vector_type(4)))  float    v4f;

#define HID 64
#define INCH 128

__device__ __forceinline__ float sigmoidf_(float x) {
    return 1.0f / (1.0f + __expf(-x));
}

// A fragment (16x32 f16, MxK) per CDNA5 layout:
// lanes 0-15 (grp0): row m=lane,  halves 0-7 = K kb..kb+7,    halves 8-15 = K kb+16..kb+23
// lanes 16-31(grp1): row m=l-16,  halves 0-7 = K kb+8..kb+15, halves 8-15 = K kb+24..kb+31
__device__ __forceinline__ v16h load_a_frag(const float* __restrict__ row, int kb, int grp, float scale) {
    const v4f* p0 = (const v4f*)(row + kb + grp * 8);
    const v4f* p1 = (const v4f*)(row + kb + 16 + grp * 8);
    v4f x0 = p0[0], x1 = p0[1];
    v4f y0 = p1[0], y1 = p1[1];
    v16h a;
#pragma unroll
    for (int i = 0; i < 4; ++i) {
        a[i]      = (_Float16)(x0[i] * scale);
        a[4 + i]  = (_Float16)(x1[i] * scale);
        a[8 + i]  = (_Float16)(y0[i] * scale);
        a[12 + i] = (_Float16)(y1[i] * scale);
    }
    return a;
}

// B fragment (32x16 f16, KxN) from LDS weight stored transposed: lwT[n*K + k].
// lane column n = lane&15; halves 0-7 = K kb+grp*8.., halves 8-15 = K kb+16+grp*8.. (2x ds_load_b128)
__device__ __forceinline__ v16h load_b_frag(const _Float16* lwT, int n, int K, int kb, int grp) {
    const _Float16* p = lwT + n * K + kb + grp * 8;
    v8h lo = *(const v8h*)p;
    v8h hi = *(const v8h*)(p + 16);
    v16h b;
#pragma unroll
    for (int i = 0; i < 8; ++i) { b[i] = lo[i]; b[8 + i] = hi[i]; }
    return b;
}

__global__ void k_zero(float* __restrict__ p, long long n) {
    long long i = (long long)blockIdx.x * blockDim.x + threadIdx.x;
    long long stride = (long long)gridDim.x * blockDim.x;
    for (; i < n; i += stride) p[i] = 0.0f;
}

// f32 [K][64] -> f16 transposed [64][K]
__global__ void k_cvt_transpose(const float* __restrict__ W, _Float16* __restrict__ WT, int K) {
    int i = blockIdx.x * blockDim.x + threadIdx.x;
    if (i < K * HID) {
        int k = i / HID, n = i % HID;
        WT[n * K + k] = (_Float16)W[i];
    }
}

__global__ void k_count(const int* __restrict__ ei, float* __restrict__ cnt, int nE) {
    int e = blockIdx.x * blockDim.x + threadIdx.x;
    if (e < nE) atomicAdd(&cnt[ei[nE + e]], 1.0f);
}

// scatter-add X[src] rows into Msum[dst]; 4 channels per thread
__global__ void k_scatter(const float* __restrict__ X, const int* __restrict__ ei,
                          float* __restrict__ Msum, int nE, int CH) {
    long long tid = (long long)blockIdx.x * blockDim.x + threadIdx.x;
    int perEdge = CH >> 2;
    long long e = tid / perEdge;
    if (e >= nE) return;
    int c = (int)(tid % perEdge) << 2;
    int s = ei[e], d = ei[nE + e];
    v4f v = *(const v4f*)(X + (long long)s * CH + c);
    float* o = Msum + (long long)d * CH + c;
    atomicAdd(o + 0, v[0]);
    atomicAdd(o + 1, v[1]);
    atomicAdd(o + 2, v[2]);
    atomicAdd(o + 3, v[3]);
}

// Out[n,64] = act( (Asum[n,:]/max(cnt,1)) @ Wl + bias + A2[n,:] @ Wr )
// K1/K2 compile-time: fully unrolled WMMA K loops, no scalar guard branches.
// 8 waves/block, each wave = 16 rows x 64 cols (4 f32 accumulators).
template <int K1, int K2, bool RELU>
__global__ void __launch_bounds__(256)
k_sage_gemm(const float* __restrict__ Asum, const float* __restrict__ cnt,
            const float* __restrict__ A2,
            const _Float16* __restrict__ WlT, const _Float16* __restrict__ WrT,
            const float* __restrict__ bias, float* __restrict__ Out,
            int nNodes) {
    __shared__ __align__(16) _Float16 lds_w[(K1 + K2) * HID];

#pragma unroll
    for (int i = threadIdx.x * 8; i < K1 * HID; i += 256 * 8)
        *(v8h*)(lds_w + i) = *(const v8h*)(WlT + i);
#pragma unroll
    for (int i = threadIdx.x * 8; i < K2 * HID; i += 256 * 8)
        *(v8h*)(lds_w + K1 * HID + i) = *(const v8h*)(WrT + i);
    __syncthreads();

    int lane = threadIdx.x & 31;
    int wave = threadIdx.x >> 5;
    int m15  = lane & 15;
    int grp  = lane >> 4;

    int tileBase = blockIdx.x * 128 + wave * 16;
    int row  = tileBase + m15;
    int rowc = row < nNodes ? row : nNodes - 1;

    float c   = cnt[rowc];
    float inv = 1.0f / (c > 1.0f ? c : 1.0f);

    v8f acc[4] = {};

    const float* r1 = Asum + (long long)rowc * K1;
#pragma unroll
    for (int kb = 0; kb < K1; kb += 32) {
        v16h a = load_a_frag(r1, kb, grp, inv);
#pragma unroll
        for (int t = 0; t < 4; ++t) {
            v16h b = load_b_frag(lds_w, 16 * t + m15, K1, kb, grp);
            acc[t] = __builtin_amdgcn_wmma_f32_16x16x32_f16(false, a, false, b,
                                                            (short)0, acc[t], false, false);
        }
    }
    const float* r2 = A2 + (long long)rowc * K2;
    const _Float16* lw2 = lds_w + K1 * HID;
#pragma unroll
    for (int kb = 0; kb < K2; kb += 32) {
        v16h a = load_a_frag(r2, kb, grp, 1.0f);
#pragma unroll
        for (int t = 0; t < 4; ++t) {
            v16h b = load_b_frag(lw2, 16 * t + m15, K2, kb, grp);
            acc[t] = __builtin_amdgcn_wmma_f32_16x16x32_f16(false, a, false, b,
                                                            (short)0, acc[t], false, false);
        }
    }

    // D layout: element j of lane -> row m = j + 8*grp, col n = 16*t + (lane&15)
#pragma unroll
    for (int t = 0; t < 4; ++t) {
        int n = 16 * t + m15;
        float bv = bias[n];
#pragma unroll
        for (int j = 0; j < 8; ++j) {
            int orow = tileBase + j + grp * 8;
            float v = acc[t][j] + bv;
            if (RELU) v = v > 0.0f ? v : 0.0f;
            if (orow < nNodes) Out[(long long)orow * HID + n] = v;
        }
    }
}

// Pair stage: gather [h2[i1]|h2[i2]] (K=128), compute q1 = feat@att_w1, q2 = feat@mlp_w1
// in one WMMA pass (combined = att*feat => shared A operand), then per-pair epilogue.
__global__ void __launch_bounds__(128)
k_pair(const float* __restrict__ h2, const int* __restrict__ pairs,
       const _Float16* __restrict__ attW1T, const _Float16* __restrict__ mlpW1T,
       const float* __restrict__ att_b1, const float* __restrict__ att_w2, const float* __restrict__ att_b2,
       const float* __restrict__ mlp_b1, const float* __restrict__ mlp_w2, const float* __restrict__ mlp_b2,
       float* __restrict__ out, int nPairs) {
    __shared__ __align__(16) char smem[32768];
    _Float16* lw = (_Float16*)smem; // weights during WMMA phase: [2][64][128] f16

#pragma unroll
    for (int i = threadIdx.x * 8; i < 2 * INCH * HID; i += 128 * 8)
        *(v8h*)(lw + i) = (i < INCH * HID) ? *(const v8h*)(attW1T + i)
                                           : *(const v8h*)(mlpW1T + (i - INCH * HID));
    __syncthreads();

    int lane = threadIdx.x & 31;
    int wave = threadIdx.x >> 5;
    int m15  = lane & 15;
    int grp  = lane >> 4;
    int tileBase = blockIdx.x * 64 + wave * 16;
    int p  = tileBase + m15;
    int pc = p < nPairs ? p : nPairs - 1;
    int i1 = pairs[2 * pc];
    int i2 = pairs[2 * pc + 1];
    const float* r1 = h2 + (long long)i1 * HID;
    const float* r2 = h2 + (long long)i2 * HID;

    v8f acc1[4] = {}, acc2[4] = {};
#pragma unroll
    for (int kb = 0; kb < 128; kb += 32) {
        const float* src = (kb < 64) ? r1 : r2;
        v16h a = load_a_frag(src, kb & 63, grp, 1.0f);
#pragma unroll
        for (int t = 0; t < 4; ++t) {
            int n = 16 * t + m15;
            v16h b1 = load_b_frag(lw, n, 128, kb, grp);
            acc1[t] = __builtin_amdgcn_wmma_f32_16x16x32_f16(false, a, false, b1,
                                                             (short)0, acc1[t], false, false);
            v16h b2 = load_b_frag(lw + INCH * HID, n, 128, kb, grp);
            acc2[t] = __builtin_amdgcn_wmma_f32_16x16x32_f16(false, a, false, b2,
                                                             (short)0, acc2[t], false, false);
        }
    }

    __syncthreads(); // all waves done with weights; reuse LDS as per-wave q tiles

    float* q = (float*)smem + wave * 2048; // q1[16][64] then q2[16][64]
#pragma unroll
    for (int t = 0; t < 4; ++t) {
        int n = 16 * t + m15;
#pragma unroll
        for (int j = 0; j < 8; ++j) {
            int m = j + grp * 8;
            q[m * HID + n]        = acc1[t][j];
            q[1024 + m * HID + n] = acc2[t][j];
        }
    }
    __syncthreads();

    if (lane < 16) {
        int m = lane;
        const float* q1 = q + m * HID;
        const float* q2 = q + 1024 + m * HID;
        float s1 = 0.0f;
#pragma unroll
        for (int n = 0; n < HID; n += 4) {
            v4f qv = *(const v4f*)(q1 + n);
            v4f bv = *(const v4f*)(att_b1 + n);
            v4f wv = *(const v4f*)(att_w2 + n);
#pragma unroll
            for (int i = 0; i < 4; ++i) {
                float v = qv[i] + bv[i];
                v = v > 0.0f ? v : 0.0f;
                s1 += v * wv[i];
            }
        }
        float att = sigmoidf_(s1 + att_b2[0]);
        float s2 = 0.0f;
#pragma unroll
        for (int n = 0; n < HID; n += 4) {
            v4f qv = *(const v4f*)(q2 + n);
            v4f bv = *(const v4f*)(mlp_b1 + n);
            v4f wv = *(const v4f*)(mlp_w2 + n);
#pragma unroll
            for (int i = 0; i < 4; ++i) {
                float v = att * qv[i] + bv[i];
                v = v > 0.0f ? v : 0.0f;
                s2 += v * wv[i];
            }
        }
        float o = sigmoidf_(s2 + mlp_b2[0]);
        int pp = tileBase + m;
        if (pp < nPairs) out[pp] = o;
    }
}

extern "C" void kernel_launch(void* const* d_in, const int* in_sizes, int n_in,
                              void* d_out, int out_size, void* d_ws, size_t ws_size,
                              hipStream_t stream) {
    (void)n_in; (void)out_size; (void)ws_size;
    const float* x     = (const float*)d_in[0];
    const int*   ei    = (const int*)d_in[1];
    const int*   pairs = (const int*)d_in[2];
    const float* w1l   = (const float*)d_in[3];
    const float* b1l   = (const float*)d_in[4];
    const float* w1r   = (const float*)d_in[5];
    const float* w2l   = (const float*)d_in[6];
    const float* b2l   = (const float*)d_in[7];
    const float* w2r   = (const float*)d_in[8];
    const float* attw1 = (const float*)d_in[9];
    const float* attb1 = (const float*)d_in[10];
    const float* attw2 = (const float*)d_in[11];
    const float* attb2 = (const float*)d_in[12];
    const float* mlpw1 = (const float*)d_in[13];
    const float* mlpb1 = (const float*)d_in[14];
    const float* mlpw2 = (const float*)d_in[15];
    const float* mlpb2 = (const float*)d_in[16];

    int nNodes = in_sizes[0] / INCH;
    int nEdges = in_sizes[1] / 2;
    int nPairs = in_sizes[2] / 2;

    // workspace layout (msg1 | cnt | msg2 contiguous for a single zeroing pass)
    float* msg1 = (float*)d_ws;
    float* cnt  = msg1 + (long long)nNodes * INCH;
    float* msg2 = cnt + nNodes;
    float* h1   = msg2 + (long long)nNodes * HID;
    float* h2   = h1 + (long long)nNodes * HID;
    _Float16* w1lT = (_Float16*)(h2 + (long long)nNodes * HID);
    _Float16* w1rT = w1lT + INCH * HID;
    _Float16* w2lT = w1rT + INCH * HID;
    _Float16* w2rT = w2lT + HID * HID;
    _Float16* attT = w2rT + HID * HID;
    _Float16* mlpT = attT + INCH * HID;

    long long zN = (long long)nNodes * INCH + nNodes + (long long)nNodes * HID;
    k_zero<<<2048, 256, 0, stream>>>(msg1, zN);

    k_cvt_transpose<<<(INCH * HID + 255) / 256, 256, 0, stream>>>(w1l, w1lT, INCH);
    k_cvt_transpose<<<(INCH * HID + 255) / 256, 256, 0, stream>>>(w1r, w1rT, INCH);
    k_cvt_transpose<<<(HID * HID + 255) / 256, 256, 0, stream>>>(w2l, w2lT, HID);
    k_cvt_transpose<<<(HID * HID + 255) / 256, 256, 0, stream>>>(w2r, w2rT, HID);
    k_cvt_transpose<<<(INCH * HID + 255) / 256, 256, 0, stream>>>(attw1, attT, INCH);
    k_cvt_transpose<<<(INCH * HID + 255) / 256, 256, 0, stream>>>(mlpw1, mlpT, INCH);

    k_count<<<(nEdges + 255) / 256, 256, 0, stream>>>(ei, cnt, nEdges);

    long long t1 = (long long)nEdges * (INCH / 4);
    k_scatter<<<(int)((t1 + 255) / 256), 256, 0, stream>>>(x, ei, msg1, nEdges, INCH);

    int gemmBlocks = (nNodes + 127) / 128;
    k_sage_gemm<INCH, INCH, true><<<gemmBlocks, 256, 0, stream>>>(msg1, cnt, x,
                                                                  w1lT, w1rT, b1l, h1, nNodes);

    long long t2 = (long long)nEdges * (HID / 4);
    k_scatter<<<(int)((t2 + 255) / 256), 256, 0, stream>>>(h1, ei, msg2, nEdges, HID);

    k_sage_gemm<HID, HID, false><<<gemmBlocks, 256, 0, stream>>>(msg2, cnt, h1,
                                                                 w2lT, w2rT, b2l, h2, nNodes);

    int pairBlocks = (nPairs + 63) / 64;
    k_pair<<<pairBlocks, 128, 0, stream>>>(h2, pairs, attT, mlpT,
                                           attb1, attw2, attb2,
                                           mlpb1, mlpw2, mlpb2,
                                           (float*)d_out, nPairs);
}
